// MultiHeadAttentionLayer_25348896981545
// MI455X (gfx1250) — matrix operand order repaired and checked by reading, not compiled, not run on hardware
//
#include <hip/hip_runtime.h>
#include <hip/hip_bf16.h>

// ---------------------------------------------------------------------------
// TUPE multi-head attention for MI455X (gfx1250), wave32, bf16 WMMA.
//   B=4, S=1024, D_MODEL=D_EMBED=1024, H=16, Dh=64
// Pipeline:
//   1) weights fp32 -> bf16
//   2) fused QKV projection (bf16 WMMA, 32x64 tiles/wave), V stored [B,H,Dh,S]
//   3) flash attention with TUPE bias + mask; K/V tiles double-buffered in LDS
//      via gfx1250 async global->LDS copies; TUPE streamed non-temporal
//   4) output projection (bf16 WMMA, 32x64 tiles/wave), fp32 NT stores
// ---------------------------------------------------------------------------

typedef __attribute__((ext_vector_type(16))) __bf16 v16bf;
typedef __attribute__((ext_vector_type(8)))  __bf16 v8bf;
typedef __attribute__((ext_vector_type(8)))  float  v8f;
typedef __attribute__((ext_vector_type(4)))  int    v4i;

#define NH 16
#define DM 1024
#define DH 64
#define BB 4
#define SS 1024

// ---- gfx1250 async global->LDS copy (guarded; fallback = synchronous copy) --
#if defined(__has_builtin)
#if __has_builtin(__builtin_amdgcn_global_load_async_to_lds_b128) && \
    __has_builtin(__builtin_amdgcn_s_wait_asynccnt)
#define USE_ASYNC_COPY 1
#endif
#endif

#if defined(USE_ASYNC_COPY)
#define WAIT_ASYNC(n) __builtin_amdgcn_s_wait_asynccnt(n)
static __device__ __forceinline__ void copy16_g2l(const __bf16* g, __bf16* l) {
  __builtin_amdgcn_global_load_async_to_lds_b128(
      (__attribute__((address_space(1))) v4i*)g,
      (__attribute__((address_space(3))) v4i*)l, 0, 0);
}
#else
#define WAIT_ASYNC(n) ((void)0)
static __device__ __forceinline__ void copy16_g2l(const __bf16* g, __bf16* l) {
  *(v8bf*)l = *(const v8bf*)g;  // synchronous 16B fallback
}
#endif

static __device__ __forceinline__ v8f wmma_bf16(v16bf a, v16bf b, v8f c) {
  // D = A(16x32 bf16) x B(32x16 bf16) + C(16x16 f32)
  return __builtin_amdgcn_wmma_f32_16x16x32_bf16(
      /*neg_a=*/false, a, /*neg_b=*/false, b,
      /*c_mod=*/(short)0, c, /*reuse_a=*/false, /*reuse_b=*/false);
}

// A-operand (16x32, 16-bit): lane l<16 holds row (l&15), K in {0..7,16..23};
// lanes 16..31 hold K in {8..15,24..31}. rowp = row base + k0.
static __device__ __forceinline__ v16bf load_a_f32(const float* rowp, int hi) {
  const float* p = rowp + hi * 8;
  v8f c0 = *(const v8f*)(p);
  v8f c1 = *(const v8f*)(p + 16);
  v16bf a;
#pragma unroll
  for (int i = 0; i < 8; ++i) { a[i] = (__bf16)c0[i]; a[8 + i] = (__bf16)c1[i]; }
  return a;
}

static __device__ __forceinline__ v16bf load_a_bf16(const __bf16* rowp, int hi) {
  const __bf16* p = rowp + hi * 8;
  v8bf c0 = *(const v8bf*)(p);
  v8bf c1 = *(const v8bf*)(p + 16);
  v16bf a;
#pragma unroll
  for (int i = 0; i < 8; ++i) { a[i] = c0[i]; a[8 + i] = c1[i]; }
  return a;
}

// B-operand (32x16, 16-bit): lane l<16 holds column (l&15), K in {0..7,16..23};
// lanes 16..31 hold K in {8..15,24..31}. A column of B is a contiguous row of src.
static __device__ __forceinline__ v16bf load_b_bf16(const __bf16* rowp, int hi) {
  return load_a_bf16(rowp, hi);
}

// ---------------------------------------------------------------------------
__global__ void cvt_f32_to_bf16(const float* __restrict__ src,
                                __bf16* __restrict__ dst, int n) {
  int i = blockIdx.x * blockDim.x + threadIdx.x;
  if (i < n) dst[i] = (__bf16)src[i];
}

// ---------------------------------------------------------------------------
// QKV projection: y[b,h,s,d] = sum_e X[b,s,e] * W[h*64+d, e]
// One wave -> 32(s) x 64(d == one full head) tile: 8 accumulators, every
// weight (B-operand) load feeds two WMMAs.
// grid = (S/32, H, 3*B); block = 32 (one wave)
__global__ void qkv_proj_kernel(const float* __restrict__ Q,
                                const float* __restrict__ K,
                                const float* __restrict__ V,
                                const __bf16* __restrict__ wq,
                                const __bf16* __restrict__ wk,
                                const __bf16* __restrict__ wv,
                                __bf16* __restrict__ qh,
                                __bf16* __restrict__ kh,
                                __bf16* __restrict__ vt) {
  const int lane = threadIdx.x;
  const int hi = lane >> 4, ln = lane & 15;
  const int s0 = blockIdx.x * 32;
  const int h = blockIdx.y;
  const int b = blockIdx.z / 3;
  const int which = blockIdx.z % 3;

  const float*  X = (which == 0) ? Q : (which == 1) ? K : V;
  const __bf16* W = (which == 0) ? wq : (which == 1) ? wk : wv;

  v8f zero = {0.f, 0.f, 0.f, 0.f, 0.f, 0.f, 0.f, 0.f};
  v8f acc[8];
#pragma unroll
  for (int n = 0; n < 8; ++n) acc[n] = zero;

  const float* xrow0 = X + ((size_t)b * SS + s0 + ln) * DM;
  const float* xrow1 = xrow0 + (size_t)16 * DM;

  for (int e0 = 0; e0 < DM; e0 += 32) {
    v16bf a0 = load_a_f32(xrow0 + e0, hi);
    v16bf a1 = load_a_f32(xrow1 + e0, hi);
#pragma unroll
    for (int n = 0; n < 4; ++n) {
      const __bf16* wrow = W + (size_t)(h * DH + n * 16 + ln) * DM + e0;
      v16bf bm = load_b_bf16(wrow, hi);
      acc[n]     = wmma_bf16(a0, bm, acc[n]);
      acc[4 + n] = wmma_bf16(a1, bm, acc[4 + n]);
    }
  }

  if (which == 2) {
    // V transposed: vt[b,h,d,s] so P*V B-operand columns are contiguous.
#pragma unroll
    for (int half = 0; half < 2; ++half) {
#pragma unroll
      for (int n = 0; n < 4; ++n) {
        int d = n * 16 + ln;
        __bf16* dst = vt + (((size_t)b * NH + h) * DH + d) * SS +
                      s0 + half * 16 + hi * 8;
#pragma unroll
        for (int r = 0; r < 8; ++r) dst[r] = (__bf16)acc[half * 4 + n][r];
      }
    }
  } else {
    __bf16* O = (which == 0) ? qh : kh;
#pragma unroll
    for (int half = 0; half < 2; ++half) {
#pragma unroll
      for (int r = 0; r < 8; ++r) {
        int row = s0 + half * 16 + r + hi * 8;
        size_t base = (((size_t)b * NH + h) * SS + row) * DH;
#pragma unroll
        for (int n = 0; n < 4; ++n)
          O[base + n * 16 + ln] = (__bf16)acc[half * 4 + n][r];
      }
    }
  }
}

// ---------------------------------------------------------------------------
// Flash attention with TUPE bias and key mask.
// One wave -> 16 q-rows x full head; K-loop in steps of 32 keys.
// K/V 4KB tiles double-buffered in LDS via async global->LDS copies.
// grid = (S/16, H, B); block = 32
__global__ void flash_attn_kernel(const __bf16* __restrict__ qh,
                                  const __bf16* __restrict__ kh,
                                  const __bf16* __restrict__ vt,
                                  const float* __restrict__ tupe,
                                  const int* __restrict__ mask,
                                  __bf16* __restrict__ concat) {
  __shared__ __bf16 kbuf[2][32 * DH];   // K tile: 32 rows x 64, 4KB each
  __shared__ __bf16 vbuf[2][DH * 32];   // V^T tile: 64 rows x 32, 4KB each
  __shared__ __bf16 pbuf[16 * 32];      // P tile: C-layout -> A-layout staging

  const int lane = threadIdx.x;
  const int hi = lane >> 4, ln = lane & 15;
  const int q0 = blockIdx.x * 16;
  const int h = blockIdx.y;
  const int b = blockIdx.z;
  const size_t bh = (size_t)b * NH + h;
  const float inv_scale = 0.08838834764831845f;  // 1/sqrt(2*Dh)

  const __bf16* qbase = qh + (bh * SS + q0 + ln) * DH;
  const __bf16* kbase = kh + bh * SS * DH;   // + kt*DH : 4KB contiguous tile
  const __bf16* vbase = vt + bh * DH * SS;   // + d*SS + kt : 64B rows

  v8f zero = {0.f, 0.f, 0.f, 0.f, 0.f, 0.f, 0.f, 0.f};
  v8f o[4];
#pragma unroll
  for (int n = 0; n < 4; ++n) o[n] = zero;
  float m_r[8], l_r[8];
#pragma unroll
  for (int r = 0; r < 8; ++r) { m_r[r] = -3.0e38f; l_r[r] = 0.f; }

  // Stage tile kt into LDS buffer p: 8+8 x 16B per lane (ASYNCcnt += 16/wave).
  auto stage = [&](int kt, int p) {
#pragma unroll
    for (int it = 0; it < 8; ++it) {  // K: 4096B contiguous
      int off = (it * 32 + lane) * 8;
      copy16_g2l(kbase + (size_t)kt * DH + off, &kbuf[p][off]);
    }
#pragma unroll
    for (int it = 0; it < 8; ++it) {  // V^T: 64 rows of 64B, stride SS
      int row = it * 8 + (lane >> 2);
      int col = (lane & 3) * 8;
      copy16_g2l(vbase + (size_t)row * SS + kt + col, &vbuf[p][row * 32 + col]);
    }
  };

  stage(0, 0);

  for (int kt = 0; kt < SS; kt += 32) {
    const int cur = (kt >> 5) & 1;
    if (kt + 32 < SS) {
      stage(kt + 32, cur ^ 1);  // prefetch next tile while waiting on this one
      WAIT_ASYNC(16);           // current tile's 16 copies have completed
    } else {
      WAIT_ASYNC(0);
    }
    __syncthreads();

    float p_s[2][8];
#pragma unroll
    for (int blk = 0; blk < 2; ++blk) {
      v8f sacc = zero;
      const __bf16* krow = &kbuf[cur][(blk * 16 + ln) * DH];
#pragma unroll
      for (int dk = 0; dk < DH; dk += 32) {
        v16bf a = load_a_bf16(qbase + dk, hi);
        sacc = wmma_bf16(a, load_b_bf16(krow + dk, hi), sacc);
      }
      const int col = kt + blk * 16 + ln;
      const int mval = mask[b * SS + col];
      const float* trow = tupe + (bh * SS + q0 + hi * 8) * (size_t)SS + col;
#pragma unroll
      for (int r = 0; r < 8; ++r) {
        // TUPE is 256MB streamed once: non-temporal, keep K/V/W in L2.
        float tv = __builtin_nontemporal_load(trow + (size_t)r * SS);
        float sv = fmaf(sacc[r], inv_scale, tv);
        p_s[blk][r] = (mval == 0) ? -1.0e9f : sv;
      }
    }

    // Online softmax per q-row (row lives on 16 lanes of one half-wave).
#pragma unroll
    for (int r = 0; r < 8; ++r) {
      float mx = fmaxf(p_s[0][r], p_s[1][r]);
#pragma unroll
      for (int off = 1; off < 16; off <<= 1)
        mx = fmaxf(mx, __shfl_xor(mx, off, 32));
      float mnew = fmaxf(m_r[r], mx);
      float scale = __expf(m_r[r] - mnew);
      float p0 = __expf(p_s[0][r] - mnew);
      float p1 = __expf(p_s[1][r] - mnew);
      float rs = p0 + p1;
#pragma unroll
      for (int off = 1; off < 16; off <<= 1) rs += __shfl_xor(rs, off, 32);
      l_r[r] = l_r[r] * scale + rs;
      m_r[r] = mnew;
#pragma unroll
      for (int n = 0; n < 4; ++n) o[n][r] *= scale;
      pbuf[(r + hi * 8) * 32 + ln]      = (__bf16)p0;
      pbuf[(r + hi * 8) * 32 + 16 + ln] = (__bf16)p1;
    }
    __syncthreads();  // order ds_store -> ds_load (single-wave WG)

    // P (16x32) as A-operand, V^T rows in LDS as contiguous B columns.
    v16bf ap = load_a_bf16(&pbuf[ln * 32], hi);
#pragma unroll
    for (int n = 0; n < 4; ++n) {
      const __bf16* vrow = &vbuf[cur][(n * 16 + ln) * 32];
      o[n] = wmma_bf16(ap, load_b_bf16(vrow, hi), o[n]);
    }
    __syncthreads();
  }

  // Normalize and store head output into concat[b, s, h*64 + d] (bf16).
#pragma unroll
  for (int r = 0; r < 8; ++r) {
    float inv = 1.0f / l_r[r];
    int row = q0 + r + hi * 8;
    size_t base = ((size_t)b * SS + row) * DM + h * DH;
#pragma unroll
    for (int n = 0; n < 4; ++n)
      concat[base + n * 16 + ln] = (__bf16)(o[n][r] * inv);
  }
}

// ---------------------------------------------------------------------------
// Output projection: out[b,s,e] = sum_m concat[b,s,m] * Wo[e,m]
// One wave -> 32(s) x 64(e) tile; NT stores (output is written once).
// grid = (S/32, D/64, B); block = 32
__global__ void out_proj_kernel(const __bf16* __restrict__ concat,
                                const __bf16* __restrict__ wo,
                                float* __restrict__ out) {
  const int lane = threadIdx.x;
  const int hi = lane >> 4, ln = lane & 15;
  const int s0 = blockIdx.x * 32;
  const int e0 = blockIdx.y * 64;
  const int b = blockIdx.z;

  v8f zero = {0.f, 0.f, 0.f, 0.f, 0.f, 0.f, 0.f, 0.f};
  v8f acc[8];
#pragma unroll
  for (int n = 0; n < 8; ++n) acc[n] = zero;

  const __bf16* arow0 = concat + ((size_t)b * SS + s0 + ln) * DM;
  const __bf16* arow1 = arow0 + (size_t)16 * DM;

  for (int m0 = 0; m0 < DM; m0 += 32) {
    v16bf a0 = load_a_bf16(arow0 + m0, hi);
    v16bf a1 = load_a_bf16(arow1 + m0, hi);
#pragma unroll
    for (int n = 0; n < 4; ++n) {
      const __bf16* wrow = wo + (size_t)(e0 + n * 16 + ln) * DM + m0;
      v16bf bm = load_b_bf16(wrow, hi);
      acc[n]     = wmma_bf16(a0, bm, acc[n]);
      acc[4 + n] = wmma_bf16(a1, bm, acc[4 + n]);
    }
  }

#pragma unroll
  for (int half = 0; half < 2; ++half) {
#pragma unroll
    for (int r = 0; r < 8; ++r) {
      int row = s0 + half * 16 + r + hi * 8;
      size_t base = ((size_t)b * SS + row) * DM + e0;
#pragma unroll
      for (int n = 0; n < 4; ++n)
        __builtin_nontemporal_store(acc[half * 4 + n][r],
                                    &out[base + n * 16 + ln]);
    }
  }
}

// ---------------------------------------------------------------------------
extern "C" void kernel_launch(void* const* d_in, const int* in_sizes, int n_in,
                              void* d_out, int out_size, void* d_ws, size_t ws_size,
                              hipStream_t stream) {
  (void)in_sizes; (void)n_in; (void)out_size; (void)ws_size;

  const float* Q    = (const float*)d_in[0];
  const float* K    = (const float*)d_in[1];
  const float* V    = (const float*)d_in[2];
  const float* TUPE = (const float*)d_in[3];
  const int*   mask = (const int*)d_in[4];
  const float* Wq   = (const float*)d_in[5];
  const float* Wk   = (const float*)d_in[6];
  const float* Wv   = (const float*)d_in[7];
  const float* Wo   = (const float*)d_in[8];

  char* ws = (char*)d_ws;
  const size_t SZ_HEAD = (size_t)BB * NH * SS * DH * sizeof(__bf16);  // 8 MiB
  const size_t SZ_W    = (size_t)DM * DM * sizeof(__bf16);            // 2 MiB

  __bf16* qh     = (__bf16*)(ws);
  __bf16* kh     = (__bf16*)(ws + 1 * SZ_HEAD);
  __bf16* vt     = (__bf16*)(ws + 2 * SZ_HEAD);
  __bf16* concat = (__bf16*)(ws + 3 * SZ_HEAD);
  __bf16* wqb    = (__bf16*)(ws + 4 * SZ_HEAD);
  __bf16* wkb    = (__bf16*)(ws + 4 * SZ_HEAD + 1 * SZ_W);
  __bf16* wvb    = (__bf16*)(ws + 4 * SZ_HEAD + 2 * SZ_W);
  __bf16* wob    = (__bf16*)(ws + 4 * SZ_HEAD + 3 * SZ_W);

  const int wn = DM * DM;
  dim3 cb(256), cg((wn + 255) / 256);
  cvt_f32_to_bf16<<<cg, cb, 0, stream>>>(Wq, wqb, wn);
  cvt_f32_to_bf16<<<cg, cb, 0, stream>>>(Wk, wkb, wn);
  cvt_f32_to_bf16<<<cg, cb, 0, stream>>>(Wv, wvb, wn);
  cvt_f32_to_bf16<<<cg, cb, 0, stream>>>(Wo, wob, wn);

  qkv_proj_kernel<<<dim3(SS / 32, NH, 3 * BB), dim3(32), 0, stream>>>(
      Q, K, V, wqb, wkb, wvb, qh, kh, vt);

  flash_attn_kernel<<<dim3(SS / 16, NH, BB), dim3(32), 0, stream>>>(
      qh, kh, vt, TUPE, mask, concat);

  out_proj_kernel<<<dim3(SS / 32, DM / 64, BB), dim3(32), 0, stream>>>(
      concat, wob, (float*)d_out);
}
